// EP_SRNNAdapter_32057635897704
// MI455X (gfx1250) — matrix operand order repaired and checked by reading, not compiled, not run on hardware
//
#include <hip/hip_runtime.h>
#include <hip/hip_bf16.h>

// ---------------------------------------------------------------------------
// SRNN forward scan on gfx1250 (MI455X).
//   v1 = ALPHA*v + z@WrecT + x@WinT - z*THR ; z1 = v1>THR ;
//   vo1 = KAPPA*vo + z1@WoutT ; logits = mean(softmax(vo), last 10 steps)
//
// Strategy: persistent kernel, 65 workgroups (64 GEMM tiles + 1 readout WG),
// device-wide barrier per timestep, WMMA f16 (hi/lo split => ~f32 accuracy,
// z is binary so exact in f16), state kept in WMMA accumulator VGPRs.
// ---------------------------------------------------------------------------

typedef __attribute__((ext_vector_type(16))) _Float16 v16h;
typedef __attribute__((ext_vector_type(8)))  _Float16 v8h;
typedef __attribute__((ext_vector_type(8)))  float    v8f;
typedef __attribute__((ext_vector_type(4)))  float    v4f;

#define ALPHA_C 0.9512294245007140f   // exp(-1/20)
#define KAPPA_C 0.9512294245007140f
#define THR_C   1.0f

constexpr int NBLK_TILE = 64;   // 2 M-tiles (128 rows) x 32 N-tiles (32 cols)
constexpr int NBLK      = 65;   // + readout workgroup
constexpr int NSTEP     = 150;  // it = 0..149 (149 v-updates, 150 publishes)
constexpr int KT_REC    = 32;   // K=1024 / 32
constexpr int KT_IN     = 22;   // K=704  / 32  (700 padded)
constexpr size_t ZPAR_HALVES = (size_t)16 * 32 * 512;  // 512 frags * 512 halves

__device__ inline v8f wmma16(v16h a, v16h b, v8f c) {
  return __builtin_amdgcn_wmma_f32_16x16x32_f16(false, a, false, b, (short)0, c,
                                                false, false);
}
__device__ inline const v16h* fragp(const _Float16* base, int frag, int lane) {
  return (const v16h*)(base + ((size_t)frag * 32 + lane) * 16);
}

// ---------------------------------------------------------------------------
// Pack a weight matrix (B operand, B[k][n] = W[n][k]) into per-fragment
// hi/lo f16 layout matching V_WMMA 16x16x32 B-matrix VGPR mapping:
// lane<16: K = 0..7,16..23 ; lane>=16: K = 8..15,24..31 ; N = nf*16 + (lane&15).
// mode 0: w_rec (zero diagonal)  mode 1: w_in (pad K>=700)  mode 2: w_out (pad N>=20)
// ---------------------------------------------------------------------------
__global__ __launch_bounds__(256)
void pack_b(const float* __restrict__ src, _Float16* __restrict__ dhi,
            _Float16* __restrict__ dlo, int nfrN, int mode) {
  const int f  = blockIdx.x;
  const int kt = f / nfrN, nf = f % nfrN;
  const int tid = threadIdx.x;
  const int l = tid >> 3, j = tid & 7;
  const int N = nf * 16 + (l & 15);
  const size_t base = ((size_t)f * 32 + l) * 16 + 2 * j;
#pragma unroll
  for (int h = 0; h < 2; ++h) {
    const int K = kt * 32 + ((l < 16) ? 0 : 8) + 2 * j + h + ((j < 4) ? 0 : 8);
    float v = 0.0f;
    if (mode == 0)      { if (N != K)  v = src[(size_t)N * 1024 + K]; }
    else if (mode == 1) { if (K < 700) v = src[(size_t)N * 700  + K]; }
    else                { if (N < 20)  v = src[(size_t)N * 1024 + K]; }
    const _Float16 hi = (_Float16)v;
    dhi[base + h] = hi;
    dlo[base + h] = (_Float16)(v - (float)hi);
  }
}

__global__ void srnn_init(unsigned int* cnt) { *cnt = 0u; }

// ---------------------------------------------------------------------------
// Persistent scan kernel.
// ---------------------------------------------------------------------------
__global__ __launch_bounds__(256)
void srnn_persistent(const float* __restrict__ X,
                     const _Float16* __restrict__ wrec_hi,
                     const _Float16* __restrict__ wrec_lo,
                     const _Float16* __restrict__ win_hi,
                     const _Float16* __restrict__ win_lo,
                     const _Float16* __restrict__ wout_hi,
                     const _Float16* __restrict__ wout_lo,
                     _Float16* __restrict__ zbuf,   // 2 parities, A-frag packed
                     unsigned int* __restrict__ cnt,
                     float* __restrict__ out) {
  __shared__ alignas(32) _Float16 zlds[8][16][32];  // per-wave C->A transpose
  __shared__ float volds[256 * 20];
  __shared__ float logits[256 * 20];

  const int bx  = blockIdx.x;
  const int tid = threadIdx.x;
  const int w   = tid >> 5;      // wave id (wave32)
  const int l   = tid & 31;
  const int lh  = l >> 4;
  const int ll  = l & 15;

  if (bx < NBLK_TILE) {
    // ---------------- GEMM tile workgroup: rows 128*mt.., cols 32*nt.. ------
    const int mt = bx >> 5;
    const int nt = bx & 31;
    const int m  = 8 * mt + w;       // global row-fragment id (0..15)
    const int nf0 = 2 * nt;          // global col-fragment id base
    v8f acc0 = {0.f,0.f,0.f,0.f,0.f,0.f,0.f,0.f};
    v8f acc1 = acc0;
    const float* xrow = X + (size_t)(16 * m + ll) * (150 * 700);

#pragma clang loop unroll(disable)
    for (int it = 0; it < NSTEP; ++it) {
      _Float16* zp = zbuf + (size_t)(it & 1) * ZPAR_HALVES;

      // ---- z_it = (v_it > THR): repack C-layout -> A-frag layout via LDS --
      float z0[8], z1[8];
#pragma unroll
      for (int i = 0; i < 8; ++i) {
        z0[i] = (acc0[i] > THR_C) ? 1.0f : 0.0f;
        z1[i] = (acc1[i] > THR_C) ? 1.0f : 0.0f;
        zlds[w][i + 8 * lh][ll]      = (_Float16)z0[i];
        zlds[w][i + 8 * lh][16 + ll] = (_Float16)z1[i];
      }
      asm volatile("s_wait_dscnt 0" ::: "memory");
      {
        const int kb0 = (l < 16) ? 0 : 8;
        v8h r0 = *(const v8h*)&zlds[w][ll][kb0];
        v8h r1 = *(const v8h*)&zlds[w][ll][kb0 + 16];
        v16h za;
#pragma unroll
        for (int i = 0; i < 8; ++i) { za[i] = r0[i]; za[8 + i] = r1[i]; }
        *(v16h*)(zp + ((size_t)(m * 32 + nt) * 32 + l) * 16) = za;
      }

      // ---- device-wide barrier (z published) -----------------------------
      __threadfence();
      __syncthreads();
      if (tid == 0) {
        __hip_atomic_fetch_add(cnt, 1u, __ATOMIC_RELEASE, __HIP_MEMORY_SCOPE_AGENT);
        const unsigned target = (unsigned)NBLK * (unsigned)(it + 1);
        while (__hip_atomic_load(cnt, __ATOMIC_ACQUIRE, __HIP_MEMORY_SCOPE_AGENT) < target)
          __builtin_amdgcn_s_sleep(2);
      }
      __syncthreads();
      __threadfence();                 // invalidate stale L0 lines before reads
      if (it == NSTEP - 1) break;

      // ---- v_{it+1} = ALPHA*v - THR*z + z@WrecT + x_it@WinT --------------
#pragma unroll
      for (int i = 0; i < 8; ++i) {
        acc0[i] = ALPHA_C * acc0[i] - THR_C * z0[i];
        acc1[i] = ALPHA_C * acc1[i] - THR_C * z1[i];
      }
      // recurrent GEMM, K = 1024, hi/lo split (exact: z is binary)
      for (int kt = 0; kt < KT_REC; ++kt) {
        v16h a = *fragp(zp, m * 32 + kt, l);
        if (kt + 1 < KT_REC) {
          __builtin_prefetch((const void*)fragp(wrec_hi, (kt + 1) * 64 + nf0, l), 0, 1);
          __builtin_prefetch((const void*)fragp(wrec_lo, (kt + 1) * 64 + nf0, l), 0, 1);
        }
        v16h bh0 = *fragp(wrec_hi, kt * 64 + nf0, l);
        v16h bl0 = *fragp(wrec_lo, kt * 64 + nf0, l);
        v16h bh1 = *fragp(wrec_hi, kt * 64 + nf0 + 1, l);
        v16h bl1 = *fragp(wrec_lo, kt * 64 + nf0 + 1, l);
        acc0 = wmma16(a, bh0, acc0);
        acc0 = wmma16(a, bl0, acc0);
        acc1 = wmma16(a, bh1, acc1);
        acc1 = wmma16(a, bl1, acc1);
      }
      // input GEMM, K = 704 (700 padded); x converted to hi/lo A-frags on the fly
      const float* xt = xrow + (size_t)it * 700;
      for (int kt = 0; kt < KT_IN; ++kt) {
        const int kb = kt * 32 + ((l < 16) ? 0 : 8);
        float f[16];
        v4f u0 = *(const v4f*)(xt + kb);
        v4f u1 = *(const v4f*)(xt + kb + 4);
#pragma unroll
        for (int i = 0; i < 4; ++i) { f[i] = u0[i]; f[4 + i] = u1[i]; }
        if (kb + 24 <= 700) {
          v4f u2 = *(const v4f*)(xt + kb + 16);
          v4f u3 = *(const v4f*)(xt + kb + 20);
#pragma unroll
          for (int i = 0; i < 4; ++i) { f[8 + i] = u2[i]; f[12 + i] = u3[i]; }
        } else {
#pragma unroll
          for (int i = 0; i < 8; ++i) {
            const int k = kb + 16 + i;
            f[8 + i] = (k < 700) ? xt[k] : 0.0f;
          }
        }
        v16h ah, al;
#pragma unroll
        for (int i = 0; i < 16; ++i) {
          const _Float16 h = (_Float16)f[i];
          ah[i] = h;
          al[i] = (_Float16)(f[i] - (float)h);
        }
        v16h bh0 = *fragp(win_hi, kt * 64 + nf0, l);
        v16h bl0 = *fragp(win_lo, kt * 64 + nf0, l);
        v16h bh1 = *fragp(win_hi, kt * 64 + nf0 + 1, l);
        v16h bl1 = *fragp(win_lo, kt * 64 + nf0 + 1, l);
        acc0 = wmma16(ah, bh0, acc0);
        acc0 = wmma16(ah, bl0, acc0);
        acc0 = wmma16(al, bh0, acc0);
        acc1 = wmma16(ah, bh1, acc1);
        acc1 = wmma16(ah, bl1, acc1);
        acc1 = wmma16(al, bh1, acc1);
      }
    }
    return;
  }

  // ---------------- readout workgroup (bx == 64): vo + softmax ------------
  for (int i = tid; i < 256 * 20; i += 256) logits[i] = 0.0f;
  __syncthreads();
  v8f vo[2][2] = {};
#pragma clang loop unroll(disable)
  for (int it = 0; it < NSTEP; ++it) {
    __syncthreads();
    if (tid == 0) {
      __hip_atomic_fetch_add(cnt, 1u, __ATOMIC_RELEASE, __HIP_MEMORY_SCOPE_AGENT);
      const unsigned target = (unsigned)NBLK * (unsigned)(it + 1);
      while (__hip_atomic_load(cnt, __ATOMIC_ACQUIRE, __HIP_MEMORY_SCOPE_AGENT) < target)
        __builtin_amdgcn_s_sleep(2);
    }
    __syncthreads();
    __threadfence();
    if (it == 0) continue;  // vo_0 == 0, outside the softmax window

    const _Float16* zp = zbuf + (size_t)(it & 1) * ZPAR_HALVES;
#pragma unroll
    for (int am = 0; am < 2; ++am)
#pragma unroll
      for (int nf = 0; nf < 2; ++nf)
#pragma unroll
        for (int i = 0; i < 8; ++i) vo[am][nf][i] *= KAPPA_C;
    for (int kt = 0; kt < KT_REC; ++kt) {
#pragma unroll
      for (int am = 0; am < 2; ++am) {
        v16h a = *fragp(zp, (2 * w + am) * 32 + kt, l);
#pragma unroll
        for (int nf = 0; nf < 2; ++nf) {
          vo[am][nf] = wmma16(a, *fragp(wout_hi, kt * 2 + nf, l), vo[am][nf]);
          vo[am][nf] = wmma16(a, *fragp(wout_lo, kt * 2 + nf, l), vo[am][nf]);
        }
      }
    }
    if (it >= NSTEP - 10) {
#pragma unroll
      for (int am = 0; am < 2; ++am)
#pragma unroll
        for (int nf = 0; nf < 2; ++nf)
#pragma unroll
          for (int i = 0; i < 8; ++i) {
            const int row = 16 * (2 * w + am) + i + 8 * lh;
            const int col = 16 * nf + ll;
            if (col < 20) volds[row * 20 + col] = vo[am][nf][i];
          }
      __syncthreads();
      {
        const int row = tid;
        float mx = volds[row * 20];
#pragma unroll
        for (int c = 1; c < 20; ++c) mx = fmaxf(mx, volds[row * 20 + c]);
        float s = 0.0f, e[20];
#pragma unroll
        for (int c = 0; c < 20; ++c) { e[c] = expf(volds[row * 20 + c] - mx); s += e[c]; }
        const float inv = 1.0f / s;
#pragma unroll
        for (int c = 0; c < 20; ++c) logits[row * 20 + c] += e[c] * inv;
      }
      __syncthreads();
    }
  }
  {
    const int row = tid;
#pragma unroll
    for (int c = 0; c < 20; ++c) out[row * 20 + c] = logits[row * 20 + c] * 0.1f;
  }
}

// ---------------------------------------------------------------------------
extern "C" void kernel_launch(void* const* d_in, const int* in_sizes, int n_in,
                              void* d_out, int out_size, void* d_ws, size_t ws_size,
                              hipStream_t stream) {
  (void)in_sizes; (void)n_in; (void)out_size; (void)ws_size;
  const float* X    = (const float*)d_in[0];   // [256,150,700]
  const float* win  = (const float*)d_in[1];   // [1024,700]
  const float* wrec = (const float*)d_in[2];   // [1024,1024]
  const float* wout = (const float*)d_in[3];   // [20,1024]

  char* ws = (char*)d_ws;
  size_t off = 0;
  auto take = [&](size_t bytes) -> void* {
    void* p = ws + off;
    off += (bytes + 255) & ~(size_t)255;
    return p;
  };
  unsigned int* cnt = (unsigned int*)take(256);
  _Float16* zbuf  = (_Float16*)take(2 * ZPAR_HALVES * 2);              // 1 MiB
  _Float16* wrech = (_Float16*)take((size_t)32 * 64 * 512 * 2);        // 2 MiB
  _Float16* wrecl = (_Float16*)take((size_t)32 * 64 * 512 * 2);
  _Float16* winh  = (_Float16*)take((size_t)22 * 64 * 512 * 2);
  _Float16* winl  = (_Float16*)take((size_t)22 * 64 * 512 * 2);
  _Float16* wouth = (_Float16*)take((size_t)32 * 2 * 512 * 2);
  _Float16* woutl = (_Float16*)take((size_t)32 * 2 * 512 * 2);
  // total workspace ~8.3 MiB

  srnn_init<<<dim3(1), dim3(1), 0, stream>>>(cnt);
  pack_b<<<dim3(32 * 64), dim3(256), 0, stream>>>(wrec, wrech, wrecl, 64, 0);
  pack_b<<<dim3(22 * 64), dim3(256), 0, stream>>>(win,  winh,  winl,  64, 1);
  pack_b<<<dim3(32 * 2),  dim3(256), 0, stream>>>(wout, wouth, woutl, 2,  2);
  srnn_persistent<<<dim3(NBLK), dim3(256), 0, stream>>>(
      X, wrech, wrecl, winh, winl, wouth, woutl, zbuf, cnt, (float*)d_out);
}
